// FlakyGreedyGIN_41686952575391
// MI455X (gfx1250) — matrix operand order repaired and checked by reading, not compiled, not run on hardware
//
#include <hip/hip_runtime.h>

#define N_NODES  50000
#define N_EDGES  400000
#define N_GRAPHS 512
#define IN_DIM   768
#define HID      64
#define BN_EPS   1e-5f

typedef __attribute__((ext_vector_type(16))) __bf16 v16bf;
typedef __attribute__((ext_vector_type(8)))  float  v8f;

union FragAB { v16bf v; __bf16 e[16]; };
union FragC  { v8f   v; float  e[8];  };

// ---------------------------------------------------------------------------
// Fused GEMM, N fixed to 64: C[M,64] = epi( bf16(A[M,K]) @ bf16(B[K,64]) )
// epi(y) = y*scale[n] + shift[n], optional ReLU.
// One wave32 -> one 16x64 strip of C (4 accumulators), so each A fragment
// feeds 4 v_wmma_f32_16x16x32_bf16 per k-step (4x A-traffic reuse).
// M must be a multiple of 16 (50000 = 3125*16) -> no store guards.
// Tile guard is wave-uniform, so EXEC is all-ones at every WMMA.
// ---------------------------------------------------------------------------
__global__ void gin_gemm_wmma64(const float* __restrict__ A,
                                const float* __restrict__ B,
                                const float* __restrict__ scale,
                                const float* __restrict__ shift,
                                float* __restrict__ C,
                                int M, int K, int relu, int numTiles)
{
    constexpr int N = 64;
    const int lane = threadIdx.x & 31;
    const int wave = threadIdx.x >> 5;
    const int tile = blockIdx.x * (blockDim.x >> 5) + wave;
    if (tile >= numTiles) return;              // wave-uniform: EXEC stays full

    const int m0   = tile << 4;
    const int half = lane >> 4;                // which K-half this lane holds
    const int mrow = m0 + (lane & 15);         // A/C row for this lane

    // Per ISA 7.12.2 16-bit A layout: element e<8  -> k = k0 + half*8 + e
    //                                 element e>=8 -> k = k0 + 16 + half*8 + (e-8)
    const float* arow = A + (size_t)mrow * K + half * 8;

    v8f acc0 = {}, acc1 = {}, acc2 = {}, acc3 = {};
    for (int k0 = 0; k0 < K; k0 += 32) {
        const float* ap = arow + k0;
        const float4 a0 = *(const float4*)(ap);
        const float4 a1 = *(const float4*)(ap + 4);
        const float4 a2 = *(const float4*)(ap + 16);
        const float4 a3 = *(const float4*)(ap + 20);

        FragAB a;
        a.e[0]  = (__bf16)a0.x; a.e[1]  = (__bf16)a0.y; a.e[2]  = (__bf16)a0.z; a.e[3]  = (__bf16)a0.w;
        a.e[4]  = (__bf16)a1.x; a.e[5]  = (__bf16)a1.y; a.e[6]  = (__bf16)a1.z; a.e[7]  = (__bf16)a1.w;
        a.e[8]  = (__bf16)a2.x; a.e[9]  = (__bf16)a2.y; a.e[10] = (__bf16)a2.z; a.e[11] = (__bf16)a2.w;
        a.e[12] = (__bf16)a3.x; a.e[13] = (__bf16)a3.y; a.e[14] = (__bf16)a3.z; a.e[15] = (__bf16)a3.w;

        // B layout: lane l holds column nt*16+(l&15); element e -> K-row same as A (table order).
        const float* bp = B + (size_t)(k0 + half * 8) * N + (lane & 15);
#pragma unroll
        for (int nt = 0; nt < 4; ++nt) {
            const float* bq = bp + nt * 16;

            FragAB b;
#pragma unroll
            for (int s = 0; s < 8; ++s) {
                b.e[s]     = (__bf16)bq[s * N];
                b.e[s + 8] = (__bf16)bq[(16 + s) * N];
            }

            if      (nt == 0) acc0 = __builtin_amdgcn_wmma_f32_16x16x32_bf16(false, a.v, false, b.v, (short)0, acc0, false, false);
            else if (nt == 1) acc1 = __builtin_amdgcn_wmma_f32_16x16x32_bf16(false, a.v, false, b.v, (short)0, acc1, false, false);
            else if (nt == 2) acc2 = __builtin_amdgcn_wmma_f32_16x16x32_bf16(false, a.v, false, b.v, (short)0, acc2, false, false);
            else              acc3 = __builtin_amdgcn_wmma_f32_16x16x32_bf16(false, a.v, false, b.v, (short)0, acc3, false, false);
        }
    }

    // Epilogue: fused bias/BN scale-shift (+ReLU).
    // C VGPR j of acc[nt] -> row m0 + half*8 + j, col nt*16 + (lane&15).
    FragC r0, r1, r2, r3;
    r0.v = acc0; r1.v = acc1; r2.v = acc2; r3.v = acc3;
    const int  cb   = lane & 15;
    float* crow = C + (size_t)(m0 + half * 8) * N;
#pragma unroll
    for (int nt = 0; nt < 4; ++nt) {
        const int   ncol = nt * 16 + cb;
        const float s = scale[ncol];
        const float t = shift[ncol];
        const FragC& r = (nt == 0) ? r0 : (nt == 1) ? r1 : (nt == 2) ? r2 : r3;
#pragma unroll
        for (int j = 0; j < 8; ++j) {
            float y = r.e[j] * s + t;
            if (relu) y = fmaxf(y, 0.0f);
            crow[j * N + ncol] = y;            // immediate row offsets, no guards
        }
    }
}

// ---------------------------------------------------------------------------
// Fold bias + BatchNorm(eval) into per-column scale/shift:
//   y = dot*s + (b - rm)*s + be   with s = g * rsqrt(rv + eps)
// ss layout: [s1a, t1a, s1b, t1b, s2a, t2a, s2b, t2b], 64 floats each.
// ---------------------------------------------------------------------------
__global__ void make_scale_shift(const float* __restrict__ b1a, const float* __restrict__ g1,
                                 const float* __restrict__ be1, const float* __restrict__ rm1,
                                 const float* __restrict__ rv1, const float* __restrict__ b1b,
                                 const float* __restrict__ b2a, const float* __restrict__ g2,
                                 const float* __restrict__ be2, const float* __restrict__ rm2,
                                 const float* __restrict__ rv2, const float* __restrict__ b2b,
                                 float* __restrict__ ss)
{
    const int n = threadIdx.x;
    if (n >= HID) return;
    const float s1 = g1[n] * rsqrtf(rv1[n] + BN_EPS);
    ss[0 * HID + n] = s1;
    ss[1 * HID + n] = (b1a[n] - rm1[n]) * s1 + be1[n];
    ss[2 * HID + n] = 1.0f;
    ss[3 * HID + n] = b1b[n];
    const float s2 = g2[n] * rsqrtf(rv2[n] + BN_EPS);
    ss[4 * HID + n] = s2;
    ss[5 * HID + n] = (b2a[n] - rm2[n]) * s2 + be2[n];
    ss[6 * HID + n] = 1.0f;
    ss[7 * HID + n] = b2b[n];
}

__global__ void copy4_k(const float4* __restrict__ src, float4* __restrict__ dst, size_t n4)
{
    const size_t i = (size_t)blockIdx.x * blockDim.x + threadIdx.x;
    if (i < n4) dst[i] = src[i];
}

__global__ void zero_k(float* __restrict__ p, size_t n)
{
    const size_t i = (size_t)blockIdx.x * blockDim.x + threadIdx.x;
    if (i < n) p[i] = 0.0f;
}

// h[dst, :] += x[src, :] — one thread per (edge, 4 features): b128 gather +
// 4 global_atomic_add_f32. This stage is the HBM roofline (~2.5 GB layer 1).
__global__ void scatter_add4_k(const float* __restrict__ xin,
                               const long long* __restrict__ srcI,
                               const long long* __restrict__ dstI,
                               float* __restrict__ h, int E, int D4)
{
    const size_t idx = (size_t)blockIdx.x * blockDim.x + threadIdx.x;
    const size_t total = (size_t)E * (size_t)D4;
    if (idx >= total) return;
    const int e  = (int)(idx / (size_t)D4);
    const int f4 = (int)(idx - (size_t)e * (size_t)D4);
    const int s = (int)srcI[e];
    const int d = (int)dstI[e];
    const float4 v = *(const float4*)(xin + ((size_t)s * D4 + f4) * 4);
    float* hp = h + ((size_t)d * D4 + f4) * 4;
    atomicAdd(hp + 0, v.x);
    atomicAdd(hp + 1, v.y);
    atomicAdd(hp + 2, v.z);
    atomicAdd(hp + 3, v.w);
}

// pool[batch[i], 0:64] += x1[i, :];  pool[batch[i], 64:128] += x2[i, :]
__global__ void pool_k(const float* __restrict__ x1, const float* __restrict__ x2,
                       const long long* __restrict__ batch, float* __restrict__ pool)
{
    const size_t idx = (size_t)blockIdx.x * blockDim.x + threadIdx.x;
    if (idx >= (size_t)N_NODES * 128) return;
    const int node = (int)(idx >> 7);
    const int f = (int)(idx & 127);
    const int g = (int)batch[node];
    const float v = (f < HID) ? x1[(size_t)node * HID + f]
                              : x2[(size_t)node * HID + (f - HID)];
    atomicAdd(&pool[g * 128 + f], v);
}

// out[g, c] = bl[c] + sum_k pool[g, k] * wl[k, c]   (512 x 2, trivially small)
__global__ void final_k(const float* __restrict__ pool, const float* __restrict__ wl,
                        const float* __restrict__ bl, float* __restrict__ out)
{
    const int idx = blockIdx.x * blockDim.x + threadIdx.x;
    if (idx >= N_GRAPHS * 2) return;
    const int g = idx >> 1, c = idx & 1;
    float acc = bl[c];
    for (int k = 0; k < 2 * HID; ++k) acc += pool[g * 128 + k] * wl[k * 2 + c];
    out[idx] = acc;
}

static inline void launch_gemm(const float* A, const float* B,
                               const float* scale, const float* shift, float* C,
                               int M, int K, int relu, hipStream_t stream)
{
    const int numTiles = M / 16;                          // M is a multiple of 16
    const int wavesPerBlock = 4;                          // 128 threads = 4 wave32
    const int blocks = (numTiles + wavesPerBlock - 1) / wavesPerBlock;
    gin_gemm_wmma64<<<blocks, wavesPerBlock * 32, 0, stream>>>(
        A, B, scale, shift, C, M, K, relu, numTiles);
}

extern "C" void kernel_launch(void* const* d_in, const int* in_sizes, int n_in,
                              void* d_out, int out_size, void* d_ws, size_t ws_size,
                              hipStream_t stream)
{
    (void)in_sizes; (void)n_in; (void)out_size; (void)ws_size;

    const float*     x     = (const float*)d_in[0];
    const long long* ei    = (const long long*)d_in[1];   // [2, E] row-major, int64
    const long long* batch = (const long long*)d_in[2];
    const float* w1a = (const float*)d_in[3];
    const float* b1a = (const float*)d_in[4];
    const float* g1  = (const float*)d_in[5];
    const float* be1 = (const float*)d_in[6];
    const float* rm1 = (const float*)d_in[7];
    const float* rv1 = (const float*)d_in[8];
    const float* w1b = (const float*)d_in[9];
    const float* b1b = (const float*)d_in[10];
    const float* w2a = (const float*)d_in[11];
    const float* b2a = (const float*)d_in[12];
    const float* g2  = (const float*)d_in[13];
    const float* be2 = (const float*)d_in[14];
    const float* rm2 = (const float*)d_in[15];
    const float* rv2 = (const float*)d_in[16];
    const float* w2b = (const float*)d_in[17];
    const float* b2b = (const float*)d_in[18];
    const float* wl  = (const float*)d_in[19];
    const float* bl  = (const float*)d_in[20];
    float* out = (float*)d_out;

    // Workspace layout (all offsets 256B-aligned)
    char* ws = (char*)d_ws;
    float* h1   = (float*)(ws + 0ull);            // 50000 x 768  (153.6 MB)
    float* t1   = (float*)(ws + 153600000ull);    // 50000 x 64
    float* x1   = (float*)(ws + 166400000ull);    // 50000 x 64
    float* h2   = (float*)(ws + 179200000ull);    // 50000 x 64
    float* t2   = (float*)(ws + 192000000ull);    // 50000 x 64
    float* x2   = (float*)(ws + 204800000ull);    // 50000 x 64
    float* pool = (float*)(ws + 217600000ull);    // 512 x 128
    float* ss   = (float*)(ws + 217862144ull);    // 8 x 64 scale/shift

    const long long* srcI = ei;                   // edge_index[0, :]
    const long long* dstI = ei + N_EDGES;         // edge_index[1, :]

    // 0) fold bias + BN into per-column scale/shift
    make_scale_shift<<<1, HID, 0, stream>>>(b1a, g1, be1, rm1, rv1, b1b,
                                            b2a, g2, be2, rm2, rv2, b2b, ss);

    // 1) h1 = x ; h1[dst] += x[src]   (layer-1 aggregation, bandwidth-dominant)
    {
        const size_t n4 = (size_t)N_NODES * IN_DIM / 4;
        copy4_k<<<(unsigned)((n4 + 255) / 256), 256, 0, stream>>>((const float4*)x, (float4*)h1, n4);
        const size_t tot = (size_t)N_EDGES * (IN_DIM / 4);
        scatter_add4_k<<<(unsigned)((tot + 255) / 256), 256, 0, stream>>>(x, srcI, dstI, h1, N_EDGES, IN_DIM / 4);
    }

    // 2) t1 = relu(BN(h1 @ w1a + b1a)) ; x1 = relu(t1 @ w1b + b1b)
    launch_gemm(h1, w1a, ss + 0 * HID, ss + 1 * HID, t1, N_NODES, IN_DIM, 1, stream);
    launch_gemm(t1, w1b, ss + 2 * HID, ss + 3 * HID, x1, N_NODES, HID,    1, stream);

    // 3) h2 = x1 ; h2[dst] += x1[src]  (layer-2 aggregation)
    {
        const size_t n4 = (size_t)N_NODES * HID / 4;
        copy4_k<<<(unsigned)((n4 + 255) / 256), 256, 0, stream>>>((const float4*)x1, (float4*)h2, n4);
        const size_t tot = (size_t)N_EDGES * (HID / 4);
        scatter_add4_k<<<(unsigned)((tot + 255) / 256), 256, 0, stream>>>(x1, srcI, dstI, h2, N_EDGES, HID / 4);
    }

    // 4) t2 = relu(BN(h2 @ w2a + b2a)) ; x2 = relu(t2 @ w2b + b2b)
    launch_gemm(h2, w2a, ss + 4 * HID, ss + 5 * HID, t2, N_NODES, HID, 1, stream);
    launch_gemm(t2, w2b, ss + 6 * HID, ss + 7 * HID, x2, N_NODES, HID, 1, stream);

    // 5) pool = segment_sum(concat(x1, x2), batch) ; out = pool @ wl + bl
    {
        const size_t np = (size_t)N_GRAPHS * 128;
        zero_k<<<(unsigned)((np + 255) / 256), 256, 0, stream>>>(pool, np);
        const size_t tot = (size_t)N_NODES * 128;
        pool_k<<<(unsigned)((tot + 255) / 256), 256, 0, stream>>>(x1, x2, batch, pool);
        final_k<<<4, 256, 0, stream>>>(pool, wl, bl, out);
    }
}
